// GraphATANode_13898514170726
// MI455X (gfx1250) — matrix operand order repaired and checked by reading, not compiled, not run on hardware
//
#include <hip/hip_runtime.h>
#include <hip/hip_fp16.h>
#include <hip/hip_bf16.h>

#define NNODES   50000
#define NEDGES   600000
#define FDIM     128
#define NCLS     40
#define NCLS_PAD 48
#define BN_EPS   1e-5f

typedef __attribute__((ext_vector_type(16))) _Float16 v16h;
typedef __attribute__((ext_vector_type(8)))  float    v8f;

// ---------------- utility kernels ----------------

__global__ void k_zero(float* p, long n) {
    long i = (long)blockIdx.x * blockDim.x + threadIdx.x;
    long stride = (long)gridDim.x * blockDim.x;
    for (; i < n; i += stride) p[i] = 0.0f;
}

// Pack W [FDIM, NOUT_SRC] (f32, row-major, K-major) into WMMA B-fragment order (f16):
// Wp[((ct*(FDIM/32) + ks)*32 + lane)*16 + e] = W[K, col]
//   half = lane>>4, idx16 = lane&15, K = ks*32 + half*16 + e, col = ct*16 + idx16
// Each lane's 16-half fragment is then 32 contiguous bytes -> b128-pair loads.
template<int NOUT_PAD, int NOUT_SRC>
__global__ void k_pack_w(const float* __restrict__ W, _Float16* __restrict__ Wp) {
    static_assert(FDIM / 32 == 4, "K-step count assumed 4");
    int i = blockIdx.x * blockDim.x + threadIdx.x;
    if (i >= NOUT_PAD * FDIM) return;
    const int e     = i & 15;
    const int lane  = (i >> 4) & 31;
    const int frag  = i >> 9;
    const int ks    = frag & 3;
    const int ct    = frag >> 2;
    const int half  = lane >> 4;
    const int idx16 = lane & 15;
    const int K     = ks * 32 + half * 16 + e;
    const int col   = ct * 16 + idx16;
    Wp[i] = (col < NOUT_SRC) ? (_Float16)W[K * NOUT_SRC + col] : (_Float16)0.0f;
}

// ---------------- graph aggregation ----------------

__global__ void k_deg(const long long* __restrict__ ei, float* __restrict__ deg) {
    int e = blockIdx.x * blockDim.x + threadIdx.x;
    if (e < NEDGES) {
        int dst = (int)ei[NEDGES + e];
        unsafeAtomicAdd(&deg[dst], 1.0f);
    }
}

__global__ void k_invdeg(float* deg) {
    int i = blockIdx.x * blockDim.x + threadIdx.x;
    if (i < NNODES) deg[i] = 1.0f / fmaxf(deg[i], 1.0f);
}

// one 32-lane group per edge; each lane moves 4 floats (float4 gather + 4 fp atomics)
__global__ void k_scatter(const long long* __restrict__ ei,
                          const float* __restrict__ X, float* __restrict__ agg) {
    long gid = (long)blockIdx.x * blockDim.x + threadIdx.x;
    long e   = gid >> 5;
    if (e >= NEDGES) return;
    int sub = (int)(gid & 31);
    int src = (int)ei[e];
    int dst = (int)ei[NEDGES + e];
    const float4 v = *(const float4*)(X + (long)src * FDIM + sub * 4);
    float* d = agg + (long)dst * FDIM + sub * 4;
    unsafeAtomicAdd(d + 0, v.x);
    unsafeAtomicAdd(d + 1, v.y);
    unsafeAtomicAdd(d + 2, v.z);
    unsafeAtomicAdd(d + 3, v.w);
}

// ---------------- WMMA GEMM: Y[16 rows] = (A*invdeg) @ W + bias ----------------
// block = (NOUT_PAD/16) waves; wave w owns column tile w; block b owns row tile b.
// Wp is pre-packed in fragment order by k_pack_w.
template<int NOUT_PAD, int NOUT_STORE, bool USE_INV>
__global__ void k_gemm_wmma(const float* __restrict__ A, const float* __restrict__ invdeg,
                            const _Float16* __restrict__ Wp, const float* __restrict__ bias,
                            float* __restrict__ Y) {
    const int wave  = threadIdx.x >> 5;
    const int lane  = threadIdx.x & 31;
    const int half  = lane >> 4;     // 0: lanes 0-15, 1: lanes 16-31
    const int idx16 = lane & 15;
    const int r0  = blockIdx.x * 16;
    const int c0  = wave * 16;
    const int row = r0 + idx16;      // A-matrix row this lane feeds
    const int col = c0 + idx16;      // C/D column this lane holds

    const float* arow = A + (long)row * FDIM;
    const float inv = USE_INV ? invdeg[row] : 1.0f;

    // C init = bias (counted once across the chained WMMAs)
    v8f acc;
    const float bv = (col < NOUT_STORE) ? bias[col] : 0.0f;
#pragma unroll
    for (int j = 0; j < 8; ++j) acc[j] = bv;

#pragma unroll
    for (int ks = 0; ks < FDIM / 32; ++ks) {
        const int k0 = ks * 32;
        // A fragment: contiguous K order (K = half*16 + e), matching packed B
        v16h a;
        const int abase = k0 + half * 16;
#pragma unroll
        for (int e = 0; e < 8; ++e) {
            a[e]     = (_Float16)(arow[abase + e]      * inv);
            a[e + 8] = (_Float16)(arow[abase + 8 + e] * inv);
        }
        // B fragment: 32 contiguous bytes per lane from the packed buffer
        const v16h b = *(const v16h*)(Wp + ((size_t)((wave * (FDIM / 32) + ks) * 32 + lane) * 16));
        acc = __builtin_amdgcn_wmma_f32_16x16x32_f16(
            /*neg_a=*/false, a, /*neg_b=*/false, b,
            /*c_mod=*/(short)0, acc, /*reuse_a=*/false, /*reuse_b=*/false);
    }

    if (col < NOUT_STORE) {
#pragma unroll
        for (int j = 0; j < 8; ++j) {  // VGPR j -> M = j + 8*half
            Y[(long)(r0 + j + 8 * half) * NOUT_STORE + col] = acc[j];
        }
    }
}

// ---------------- BN stats + apply ----------------

__global__ void k_colstats(const float* __restrict__ Y,
                           float* __restrict__ sums, float* __restrict__ sumsq) {
    __shared__ float s1[256], s2[256];
    const int col  = threadIdx.x & (FDIM - 1);
    const int part = threadIdx.x >> 7;          // 0 or 1
    float a = 0.0f, b = 0.0f;
    for (int row = blockIdx.x * 2 + part; row < NNODES; row += gridDim.x * 2) {
        float v = Y[(long)row * FDIM + col];
        a += v; b += v * v;
    }
    s1[threadIdx.x] = a; s2[threadIdx.x] = b;
    __syncthreads();
    if (part == 0) {
        a += s1[threadIdx.x + 128];
        b += s2[threadIdx.x + 128];
        unsafeAtomicAdd(&sums[col], a);
        unsafeAtomicAdd(&sumsq[col], b);
    }
}

__global__ void k_bnrelu(float* __restrict__ Y,
                         const float* __restrict__ sums, const float* __restrict__ sumsq,
                         const float* __restrict__ g, const float* __restrict__ be) {
    long i = (long)blockIdx.x * blockDim.x + threadIdx.x;
    if (i >= (long)NNODES * FDIM) return;
    const int col = (int)(i & (FDIM - 1));
    const float mean = sums[col] * (1.0f / NNODES);
    const float var  = sumsq[col] * (1.0f / NNODES) - mean * mean;
    const float sc   = g[col] * rsqrtf(var + BN_EPS);
    const float v    = (Y[i] - mean) * sc + be[col];
    Y[i] = fmaxf(v, 0.0f);
}

// ---------------- row-wise log-softmax over 40 cols, one wave32 per row ----------------

__global__ void k_logsoftmax(float* __restrict__ out) {
    const int warp = (int)(((long)blockIdx.x * blockDim.x + threadIdx.x) >> 5);
    const int lane = threadIdx.x & 31;
    if (warp >= NNODES) return;
    float* row = out + (long)warp * NCLS;
    const bool has2 = (lane + 32) < NCLS;
    float v0 = row[lane];
    float v1 = has2 ? row[lane + 32] : -3.0e38f;
    float m = fmaxf(v0, v1);
#pragma unroll
    for (int off = 16; off > 0; off >>= 1) m = fmaxf(m, __shfl_xor(m, off, 32));
    float s = expf(v0 - m) + (has2 ? expf(v1 - m) : 0.0f);
#pragma unroll
    for (int off = 16; off > 0; off >>= 1) s += __shfl_xor(s, off, 32);
    const float lse = m + logf(s);
    row[lane] = v0 - lse;
    if (has2) row[lane + 32] = v1 - lse;
}

// ---------------- launch ----------------

extern "C" void kernel_launch(void* const* d_in, const int* in_sizes, int n_in,
                              void* d_out, int out_size, void* d_ws, size_t ws_size,
                              hipStream_t stream) {
    const float*     x   = (const float*)d_in[0];
    const long long* ei  = (const long long*)d_in[1];   // int64 [2, E]
    const float*     W0  = (const float*)d_in[2];
    const float*     b0  = (const float*)d_in[3];
    const float*     g0  = (const float*)d_in[4];
    const float*     be0 = (const float*)d_in[5];
    const float*     W1  = (const float*)d_in[6];
    const float*     b1  = (const float*)d_in[7];
    const float*     g1  = (const float*)d_in[8];
    const float*     be1 = (const float*)d_in[9];
    const float*     Wc  = (const float*)d_in[10];
    const float*     bc  = (const float*)d_in[11];
    float* out = (float*)d_out;

    char* ws = (char*)d_ws;
    size_t off = 0;
    auto alloc = [&](size_t bytes) -> void* {
        void* p = ws + off;
        off += (bytes + 255) & ~(size_t)255;
        return p;
    };
    float*    agg   = (float*)alloc((size_t)NNODES * FDIM * 4);
    float*    ybuf  = (float*)alloc((size_t)NNODES * FDIM * 4);
    float*    deg   = (float*)alloc((size_t)NNODES * 4);
    _Float16* w16   = (_Float16*)alloc((size_t)FDIM * FDIM * 2);
    _Float16* wc16  = (_Float16*)alloc((size_t)FDIM * NCLS_PAD * 2);
    float*    sums  = (float*)alloc(FDIM * 4);
    float*    sumsq = (float*)alloc(FDIM * 4);

    const int scatterBlocks = (NEDGES * 32) / 256;            // 75000 exact
    const int tiles = NNODES / 16;                            // 3125 exact
    const long elems = (long)NNODES * FDIM;
    const int bnBlocks = (int)((elems + 255) / 256);

    // ---- degree (shared by both layers) ----
    k_zero<<<256, 256, 0, stream>>>(deg, NNODES);
    k_deg<<<(NEDGES + 255) / 256, 256, 0, stream>>>(ei, deg);
    k_invdeg<<<(NNODES + 255) / 256, 256, 0, stream>>>(deg);

    // ---- layer 0: scatter(x) -> gemm W0 -> BN -> ReLU ----
    k_zero<<<2048, 256, 0, stream>>>(agg, elems);
    k_scatter<<<scatterBlocks, 256, 0, stream>>>(ei, x, agg);
    k_pack_w<FDIM, FDIM><<<(FDIM * FDIM + 255) / 256, 256, 0, stream>>>(W0, w16);
    k_gemm_wmma<FDIM, FDIM, true><<<tiles, 256, 0, stream>>>(agg, deg, w16, b0, ybuf);
    k_zero<<<1, 256, 0, stream>>>(sums, FDIM);
    k_zero<<<1, 256, 0, stream>>>(sumsq, FDIM);
    k_colstats<<<128, 256, 0, stream>>>(ybuf, sums, sumsq);
    k_bnrelu<<<bnBlocks, 256, 0, stream>>>(ybuf, sums, sumsq, g0, be0);

    // ---- layer 1: scatter(h) -> gemm W1 -> BN -> ReLU ----
    k_zero<<<2048, 256, 0, stream>>>(agg, elems);
    k_scatter<<<scatterBlocks, 256, 0, stream>>>(ei, ybuf, agg);
    k_pack_w<FDIM, FDIM><<<(FDIM * FDIM + 255) / 256, 256, 0, stream>>>(W1, w16);
    k_gemm_wmma<FDIM, FDIM, true><<<tiles, 256, 0, stream>>>(agg, deg, w16, b1, ybuf);
    k_zero<<<1, 256, 0, stream>>>(sums, FDIM);
    k_zero<<<1, 256, 0, stream>>>(sumsq, FDIM);
    k_colstats<<<128, 256, 0, stream>>>(ybuf, sums, sumsq);
    k_bnrelu<<<bnBlocks, 256, 0, stream>>>(ybuf, sums, sumsq, g1, be1);

    // ---- classifier + log_softmax (logits written straight into d_out) ----
    k_pack_w<NCLS_PAD, NCLS><<<(FDIM * NCLS_PAD + 255) / 256, 256, 0, stream>>>(Wc, wc16);
    k_gemm_wmma<NCLS_PAD, NCLS, false><<<tiles, 96, 0, stream>>>(ybuf, nullptr, wc16, bc, out);
    k_logsoftmax<<<(NNODES * 32) / 256, 256, 0, stream>>>(out);
}
